// DrugCellNN_17308718203006
// MI455X (gfx1250) — compile-verified, matrix operand order checked by reading, not run on hardware
//
#include <hip/hip_runtime.h>
#include <hip/hip_bf16.h>
#include <math.h>

// ---------------- problem constants ----------------
#define B_  2048
#define G_  3008
#define F_  3
#define H_  6
#define T0_ 1500
#define K_  4
#define T1_ 400
#define C1_ 4
#define T2_ 80
#define C2_ 5
#define N_   (T0_ * K_)            // 6000
#define OUTC (T0_ + T1_ + T2_ + 2) // 1982

typedef _Float16 half_t;
typedef __attribute__((ext_vector_type(4)))  float    vf4;
typedef __attribute__((ext_vector_type(4)))  _Float16 vh4;
typedef __attribute__((ext_vector_type(8)))  _Float16 v8h;
typedef __attribute__((ext_vector_type(16))) _Float16 v16h;
typedef __attribute__((ext_vector_type(8)))  float    v8f;

// ---------------- workspace layout ----------------
static constexpr size_t align256(size_t x) { return (x + 255) & ~size_t(255); }
static constexpr size_t OFF_GE  = 0;                                   // gene_emb f16 [B][G]
static constexpr size_t SZ_GE   = (size_t)B_ * G_ * 2;
static constexpr size_t OFF_WH  = align256(OFF_GE + SZ_GE);            // direct_w f16 [N][G]
static constexpr size_t SZ_WH   = (size_t)N_ * G_ * 2;
static constexpr size_t OFF_DIR = align256(OFF_WH + SZ_WH);            // direct f32 [B][N]
static constexpr size_t SZ_DIR  = (size_t)B_ * N_ * 4;
static constexpr size_t OFF_E0  = align256(OFF_DIR + SZ_DIR);          // emb0 f32 [B][T0][H]
static constexpr size_t SZ_E0   = (size_t)B_ * T0_ * H_ * 4;
static constexpr size_t OFF_E1  = align256(OFF_E0 + SZ_E0);            // emb1 f32 [B][T1][H]
static constexpr size_t SZ_E1   = (size_t)B_ * T1_ * H_ * 4;
static constexpr size_t OFF_E2  = align256(OFF_E1 + SZ_E1);            // emb2 f32 [B][T2][H]
static constexpr size_t SZ_E2   = (size_t)B_ * T2_ * H_ * 4;

// ---------------- gene embedding: tanh(x·w + b)*s + t -> f16 ----------------
__global__ void k_gene(const float* __restrict__ x, const float* __restrict__ gw,
                       const float* __restrict__ gb, const float* __restrict__ gs,
                       const float* __restrict__ gt, half_t* __restrict__ ge)
{
    int i = blockIdx.x * blockDim.x + threadIdx.x;
    if (i >= B_ * G_) return;
    int g = i % G_;
    const float* xp = x + (size_t)i * F_;
    float d = xp[0] * gw[g * F_ + 0] + xp[1] * gw[g * F_ + 1] + xp[2] * gw[g * F_ + 2] + gb[g];
    ge[i] = (half_t)(tanhf(d) * gs[g] + gt[g]);
}

// ---------------- f32 -> f16 weight conversion (4-wide) ----------------
__global__ void k_cvt(const vf4* __restrict__ src, vh4* __restrict__ dst, int n4)
{
    int i = blockIdx.x * blockDim.x + threadIdx.x;
    if (i >= n4) return;
    dst[i] = __builtin_convertvector(src[i], vh4);
}

// ---------------- WMMA GEMM helpers ----------------
__device__ __forceinline__ void load_frags(const half_t* __restrict__ aBase,
                                           const half_t* __restrict__ wBase,
                                           int k0, v16h (&af)[4], v16h (&bf)[3])
{
#pragma unroll
    for (int i = 0; i < 4; ++i) {
        const half_t* p = aBase + (size_t)i * 16 * G_ + k0;
        v8h lo = *(const v8h*)p;          // K = kof .. kof+7
        v8h hi = *(const v8h*)(p + 16);   // K = 16+kof .. 16+kof+7
        af[i] = __builtin_shufflevector(lo, hi, 0,1,2,3,4,5,6,7,8,9,10,11,12,13,14,15);
    }
#pragma unroll
    for (int j = 0; j < 3; ++j) {
        const half_t* p = wBase + (size_t)j * 16 * G_ + k0;
        v8h lo = *(const v8h*)p;
        v8h hi = *(const v8h*)(p + 16);
        bf[j] = __builtin_shufflevector(lo, hi, 0,1,2,3,4,5,6,7,8,9,10,11,12,13,14,15);
    }
}

__device__ __forceinline__ void mma_tile(v8f (&acc)[4][3], const v16h (&af)[4], const v16h (&bf)[3])
{
#pragma unroll
    for (int i = 0; i < 4; ++i)
#pragma unroll
        for (int j = 0; j < 3; ++j)
            acc[i][j] = __builtin_amdgcn_wmma_f32_16x16x32_f16(
                false, af[i], false, bf[j], (short)0, acc[i][j], false, false);
}

// ---------------- main GEMM: D[B,N] = A[B,G] * W[N,G]^T + bias, via WMMA f16 ----------------
// One wave (32 threads) computes a 64x48 output tile: 4 A-frags x 3 B-frags,
// with double-buffered operand fragments (K unrolled by 64) to hide L2 latency.
__global__ __launch_bounds__(32) void k_gemm_wmma(
    const half_t* __restrict__ A,    // [B_][G_] f16
    const half_t* __restrict__ W,    // [N_][G_] f16 (direct_w rows; B columns are contiguous)
    const float*  __restrict__ bias, // [N_]
    float* __restrict__ D)           // [B_][N_]
{
    const int lane = threadIdx.x & 31;
    const int lr   = lane & 15;          // row (A) / col (B) within fragment
    const int kof  = (lane >> 4) * 8;    // half-wave K offset (0 or 8)
    const int n0 = blockIdx.x * 48;
    const int m0 = blockIdx.y * 64;

    v8f acc[4][3] = {};

    const half_t* aBase = A + (size_t)(m0 + lr) * G_ + kof;
    const half_t* wBase = W + (size_t)(n0 + lr) * G_ + kof;

    v16h afA[4], bfA[3], afB[4], bfB[3];
    load_frags(aBase, wBase, 0, afA, bfA);

    // G_ = 3008 = 47 * 64: two 32-deep chunks per iteration, ping-pong buffers.
    for (int k0 = 0; k0 < G_; k0 += 64) {
        load_frags(aBase, wBase, k0 + 32, afB, bfB);   // prefetch chunk k0+32
        mma_tile(acc, afA, bfA);                       // 12 WMMAs on chunk k0
        if (k0 + 64 < G_)
            load_frags(aBase, wBase, k0 + 64, afA, bfA); // prefetch chunk k0+64
        mma_tile(acc, afB, bfB);                       // 12 WMMAs on chunk k0+32
    }

    // C/D layout: VGPR r holds M = r + (lane>=16 ? 8 : 0), N = lane&15
#pragma unroll
    for (int j = 0; j < 3; ++j) {
        const int col = n0 + j * 16 + lr;
        const float bc = bias[col];
#pragma unroll
        for (int i = 0; i < 4; ++i) {
            const int rowBase = m0 + i * 16 + kof;
#pragma unroll
            for (int r = 0; r < 8; ++r) {
                D[(size_t)(rowBase + r) * N_ + col] = acc[i][j][r] + bc;
            }
        }
    }
}

// ---------------- leaf term block: per-(b,t) K=4 -> H=6 + aux head ----------------
__global__ void k_term0(const float* __restrict__ direct,
                        const float* __restrict__ w0, const float* __restrict__ b0,
                        const float* __restrict__ s0, const float* __restrict__ t0v,
                        const float* __restrict__ aw1, const float* __restrict__ ab1,
                        const float* __restrict__ aw2, const float* __restrict__ ab2,
                        float* __restrict__ emb0, float* __restrict__ out)
{
    int i = blockIdx.x * blockDim.x + threadIdx.x;
    if (i >= B_ * T0_) return;
    int t = i % T0_;
    int b = i / T0_;
    float in[K_];
    const float* dp = direct + (size_t)b * N_ + t * K_;
#pragma unroll
    for (int k = 0; k < K_; ++k) in[k] = dp[k];
    float e[H_];
    float a1 = ab1[t];
#pragma unroll
    for (int j = 0; j < H_; ++j) {
        const float* wp = w0 + ((size_t)t * H_ + j) * K_;
        float h = b0[t * H_ + j];
#pragma unroll
        for (int k = 0; k < K_; ++k) h += in[k] * wp[k];
        float ee = tanhf(h) * s0[t * H_ + j] + t0v[t * H_ + j];
        e[j] = ee;
        a1 += ee * aw1[t * H_ + j];
    }
    float* ep = emb0 + (size_t)i * H_;
#pragma unroll
    for (int j = 0; j < H_; ++j) ep[j] = e[j];
    out[(size_t)b * OUTC + t] = tanhf(a1) * aw2[t] + ab2[t];
}

// ---------------- DAG level: gather C children, C*H -> H + aux head ----------------
template<int T, int C, int TP, int COLOFF>
__global__ void k_level(const float* __restrict__ prev,  // [B][TP][H]
                        const int* __restrict__ cidx,    // [T][C]
                        const float* __restrict__ w,     // [T][H][C*H]
                        const float* __restrict__ bvec,  // [T][H]
                        const float* __restrict__ s, const float* __restrict__ tt,
                        const float* __restrict__ aw1, const float* __restrict__ ab1,
                        const float* __restrict__ aw2, const float* __restrict__ ab2,
                        float* __restrict__ emb, float* __restrict__ out)
{
    int i = blockIdx.x * blockDim.x + threadIdx.x;
    if (i >= B_ * T) return;
    int t = i % T;
    int b = i / T;
    float in[C * H_];
#pragma unroll
    for (int c = 0; c < C; ++c) {
        int ch = cidx[t * C + c];
        const float* pp = prev + ((size_t)b * TP + ch) * H_;
#pragma unroll
        for (int j = 0; j < H_; ++j) in[c * H_ + j] = pp[j];
    }
    float e[H_];
    float a1 = ab1[t];
#pragma unroll
    for (int j = 0; j < H_; ++j) {
        const float* wp = w + ((size_t)t * H_ + j) * (C * H_);
        float h = bvec[t * H_ + j];
#pragma unroll
        for (int k = 0; k < C * H_; ++k) h += in[k] * wp[k];
        float ee = tanhf(h) * s[t * H_ + j] + tt[t * H_ + j];
        e[j] = ee;
        a1 += ee * aw1[t * H_ + j];
    }
    float* ep = emb + (size_t)i * H_;
#pragma unroll
    for (int j = 0; j < H_; ++j) ep[j] = e[j];
    out[(size_t)b * OUTC + COLOFF + t] = tanhf(a1) * aw2[t] + ab2[t];
}

// ---------------- root term + aux_r + final ----------------
__global__ void k_root(const float* __restrict__ emb2,   // [B][T2*H]
                       const float* __restrict__ wr,     // [H][T2*H]
                       const float* __restrict__ br,
                       const float* __restrict__ bs, const float* __restrict__ bt,
                       const float* __restrict__ arw1, const float* __restrict__ arb1,
                       const float* __restrict__ arw2, const float* __restrict__ arb2,
                       const float* __restrict__ fw, const float* __restrict__ fb,
                       const float* __restrict__ fow, const float* __restrict__ fob,
                       float* __restrict__ out)
{
    int b = blockIdx.x * blockDim.x + threadIdx.x;
    if (b >= B_) return;
    const float* inr = emb2 + (size_t)b * (T2_ * H_);
    float er[H_];
#pragma unroll
    for (int j = 0; j < H_; ++j) {
        const float* wp = wr + (size_t)j * (T2_ * H_);
        float h = br[j];
        for (int k = 0; k < T2_ * H_; ++k) h += inr[k] * wp[k];
        er[j] = tanhf(h) * bs[j] + bt[j];
    }
    float a = arb1[0], f = fb[0];
#pragma unroll
    for (int j = 0; j < H_; ++j) { a += er[j] * arw1[j]; f += er[j] * fw[j]; }
    out[(size_t)b * OUTC + (T0_ + T1_ + T2_)]     = tanhf(a) * arw2[0] + arb2[0];
    out[(size_t)b * OUTC + (T0_ + T1_ + T2_ + 1)] = tanhf(f) * fow[0] + fob[0];
}

extern "C" void kernel_launch(void* const* d_in, const int* in_sizes, int n_in,
                              void* d_out, int out_size, void* d_ws, size_t ws_size,
                              hipStream_t stream)
{
    const float* x         = (const float*)d_in[0];
    const float* gene_w    = (const float*)d_in[1];
    const float* gene_b    = (const float*)d_in[2];
    const float* gene_bn_s = (const float*)d_in[3];
    const float* gene_bn_t = (const float*)d_in[4];
    const float* direct_w  = (const float*)d_in[5];
    const float* direct_b  = (const float*)d_in[6];
    const float* w0      = (const float*)d_in[7];
    const float* b0      = (const float*)d_in[8];
    const float* bn0_s   = (const float*)d_in[9];
    const float* bn0_t   = (const float*)d_in[10];
    const float* aux0_w1 = (const float*)d_in[11];
    const float* aux0_b1 = (const float*)d_in[12];
    const float* aux0_w2 = (const float*)d_in[13];
    const float* aux0_b2 = (const float*)d_in[14];
    const float* w1      = (const float*)d_in[15];
    const float* b1      = (const float*)d_in[16];
    const float* bn1_s   = (const float*)d_in[17];
    const float* bn1_t   = (const float*)d_in[18];
    const float* aux1_w1 = (const float*)d_in[19];
    const float* aux1_b1 = (const float*)d_in[20];
    const float* aux1_w2 = (const float*)d_in[21];
    const float* aux1_b2 = (const float*)d_in[22];
    const float* w2      = (const float*)d_in[23];
    const float* b2      = (const float*)d_in[24];
    const float* bn2_s   = (const float*)d_in[25];
    const float* bn2_t   = (const float*)d_in[26];
    const float* aux2_w1 = (const float*)d_in[27];
    const float* aux2_b1 = (const float*)d_in[28];
    const float* aux2_w2 = (const float*)d_in[29];
    const float* aux2_b2 = (const float*)d_in[30];
    const float* wr      = (const float*)d_in[31];
    const float* br      = (const float*)d_in[32];
    const float* bnr_s   = (const float*)d_in[33];
    const float* bnr_t   = (const float*)d_in[34];
    const float* auxr_w1 = (const float*)d_in[35];
    const float* auxr_b1 = (const float*)d_in[36];
    const float* auxr_w2 = (const float*)d_in[37];
    const float* auxr_b2 = (const float*)d_in[38];
    const float* final_w  = (const float*)d_in[39];
    const float* final_b  = (const float*)d_in[40];
    const float* final_ow = (const float*)d_in[41];
    const float* final_ob = (const float*)d_in[42];
    const int* child_idx1 = (const int*)d_in[43];
    const int* child_idx2 = (const int*)d_in[44];

    char* ws = (char*)d_ws;
    half_t* ge  = (half_t*)(ws + OFF_GE);
    half_t* wh  = (half_t*)(ws + OFF_WH);
    float*  dir = (float*)(ws + OFF_DIR);
    float*  e0  = (float*)(ws + OFF_E0);
    float*  e1  = (float*)(ws + OFF_E1);
    float*  e2  = (float*)(ws + OFF_E2);
    float*  out = (float*)d_out;

    // 1. gene embedding -> f16
    {
        int total = B_ * G_;
        k_gene<<<(total + 255) / 256, 256, 0, stream>>>(x, gene_w, gene_b, gene_bn_s, gene_bn_t, ge);
    }
    // 2. direct_w f32 -> f16
    {
        int n4 = (N_ * G_) / 4;
        k_cvt<<<(n4 + 255) / 256, 256, 0, stream>>>((const vf4*)direct_w, (vh4*)wh, n4);
    }
    // 3. big WMMA GEMM: [2048,3008] x [3008,6000] + bias
    {
        dim3 grid(N_ / 48, B_ / 64);   // 125 x 32 waves
        k_gemm_wmma<<<grid, 32, 0, stream>>>(ge, wh, direct_b, dir);
    }
    // 4. leaf term block
    {
        int total = B_ * T0_;
        k_term0<<<(total + 255) / 256, 256, 0, stream>>>(dir, w0, b0, bn0_s, bn0_t,
                                                         aux0_w1, aux0_b1, aux0_w2, aux0_b2,
                                                         e0, out);
    }
    // 5. level 1
    {
        int total = B_ * T1_;
        k_level<T1_, C1_, T0_, T0_><<<(total + 255) / 256, 256, 0, stream>>>(
            e0, child_idx1, w1, b1, bn1_s, bn1_t, aux1_w1, aux1_b1, aux1_w2, aux1_b2, e1, out);
    }
    // 6. level 2
    {
        int total = B_ * T2_;
        k_level<T2_, C2_, T1_, T0_ + T1_><<<(total + 255) / 256, 256, 0, stream>>>(
            e1, child_idx2, w2, b2, bn2_s, bn2_t, aux2_w1, aux2_b1, aux2_w2, aux2_b2, e2, out);
    }
    // 7. root + final
    {
        k_root<<<(B_ + 255) / 256, 256, 0, stream>>>(e2, wr, br, bnr_s, bnr_t,
                                                     auxr_w1, auxr_b1, auxr_w2, auxr_b2,
                                                     final_w, final_b, final_ow, final_ob, out);
    }
    (void)in_sizes; (void)n_in; (void)out_size; (void)ws_size;
}
